// TransformerDecoder_27178553049203
// MI455X (gfx1250) — matrix operand order repaired and checked by reading, not compiled, not run on hardware
//
#include <hip/hip_runtime.h>
#include <math.h>

// Problem constants (match reference)
#define B_  2
#define S_  1024
#define D_  512
#define H_  8
#define L_  4
#define V_  32000
#define DK_ 64

typedef __attribute__((ext_vector_type(16))) _Float16 v16h;
typedef __attribute__((ext_vector_type(8)))  _Float16 v8h;
typedef __attribute__((ext_vector_type(8)))  float    v8f;

// ---------------------------------------------------------------------------
// Generic batched GEMM, C = alpha * A*B (+bias, +relu), fp32 in/out,
// f16 WMMA (v_wmma_f32_16x16x32_f16) inner product, wave32, LDS-staged tiles.
//   BT = false: B is [K,N] row-major (NN).  BT = true: B is [N,K] row-major (NT).
//   EPI: 0 none, 1 bias+relu, 2 bias.
// Block = 256 threads = 8 waves arranged 4(M) x 2(N); wave tile 16x32;
// block tile 64x64, K-step 32. A/B tiles staged once per block into LDS as
// f16 (converted during staging), fragments read as aligned 16B LDS vectors.
// Requires M%64==0, N%64==0, K%32==0 (true for all shapes here).
// Batch: z0 = blockIdx.z / zdiv, z1 = blockIdx.z % zdiv; pointer offsets
// A += z0*sA0 + z1*sA1, etc.
// ---------------------------------------------------------------------------
template <bool BT, int EPI>
__global__ __launch_bounds__(256) void gemm_wmma(
    const float* __restrict__ A, const float* __restrict__ Bm,
    float* __restrict__ C, const float* __restrict__ bias,
    int M, int N, int K, int lda, int ldb, int ldc,
    int zdiv, long sA0, long sA1, long sB0, long sB1, long sC0, long sC1,
    float alpha)
{
  // Padded LDS tiles: 40 halves (80 B) row stride -> keeps 16B alignment for
  // vector reads and spreads consecutive rows across banks.
  __shared__ __align__(16) _Float16 As[64][40];  // [m][k]
  __shared__ __align__(16) _Float16 Bs[64][40];  // [n][k]

  const int z  = blockIdx.z;
  const int z0 = z / zdiv;
  const int z1 = z % zdiv;
  A  += z0 * sA0 + z1 * sA1;
  Bm += z0 * sB0 + z1 * sB1;
  C  += z0 * sC0 + z1 * sC1;

  const int tid  = threadIdx.x;
  const int wid  = tid >> 5;           // wave id 0..7
  const int lane = tid & 31;
  const int hh   = lane >> 4;          // lane half (0 or 1)
  const int l16  = lane & 15;

  const int mwl = (wid >> 1) * 16;     // wave M base within block tile
  const int nwl = (wid & 1) * 32;      // wave N base within block tile
  const int mB  = blockIdx.y * 64;     // block M base
  const int nB  = blockIdx.x * 64;     // block N base

  // Cooperative staging coordinates (256 threads, 2048 f32 per tile, 8 each):
  const int arow = tid >> 2;           // 0..63  (A row / BT row)
  const int akc  = (tid & 3) << 3;     // 0,8,16,24 (k segment of 8)
  const int krow = tid >> 3;           // 0..31  (NN: B k-row)
  const int nseg = (tid & 7) << 3;     // 0..56  (NN: B n segment of 8)

  v8f acc0 = {};
  v8f acc1 = {};

  for (int k0 = 0; k0 < K; k0 += 32) {
    // ---------------- stage A tile (64x32 f32 -> f16 LDS [m][k]) -----------
    {
      const float4* ag =
          (const float4*)(A + (long)(mB + arow) * lda + k0 + akc);
      float4 a0 = ag[0];
      float4 a1 = ag[1];
      _Float16* d = &As[arow][akc];
      d[0] = (_Float16)a0.x; d[1] = (_Float16)a0.y;
      d[2] = (_Float16)a0.z; d[3] = (_Float16)a0.w;
      d[4] = (_Float16)a1.x; d[5] = (_Float16)a1.y;
      d[6] = (_Float16)a1.z; d[7] = (_Float16)a1.w;
    }
    // ---------------- stage B tile -> f16 LDS [n][k] -----------------------
    if constexpr (BT) {
      const float4* bg =
          (const float4*)(Bm + (long)(nB + arow) * ldb + k0 + akc);
      float4 b0 = bg[0];
      float4 b1 = bg[1];
      _Float16* d = &Bs[arow][akc];
      d[0] = (_Float16)b0.x; d[1] = (_Float16)b0.y;
      d[2] = (_Float16)b0.z; d[3] = (_Float16)b0.w;
      d[4] = (_Float16)b1.x; d[5] = (_Float16)b1.y;
      d[6] = (_Float16)b1.z; d[7] = (_Float16)b1.w;
    } else {
      const float4* bg =
          (const float4*)(Bm + (long)(k0 + krow) * ldb + nB + nseg);
      float4 b0 = bg[0];
      float4 b1 = bg[1];
      Bs[nseg + 0][krow] = (_Float16)b0.x;
      Bs[nseg + 1][krow] = (_Float16)b0.y;
      Bs[nseg + 2][krow] = (_Float16)b0.z;
      Bs[nseg + 3][krow] = (_Float16)b0.w;
      Bs[nseg + 4][krow] = (_Float16)b1.x;
      Bs[nseg + 5][krow] = (_Float16)b1.y;
      Bs[nseg + 6][krow] = (_Float16)b1.z;
      Bs[nseg + 7][krow] = (_Float16)b1.w;
    }
    // Prefetch next K-tile into cache while this one computes
    // (global_prefetch_b8 on gfx1250).
    if (k0 + 32 < K) {
      __builtin_prefetch(
          (const void*)(A + (long)(mB + arow) * lda + k0 + 32 + akc), 0, 1);
      if constexpr (BT) {
        __builtin_prefetch(
            (const void*)(Bm + (long)(nB + arow) * ldb + k0 + 32 + akc), 0, 1);
      } else {
        __builtin_prefetch(
            (const void*)(Bm + (long)(k0 + 32 + krow) * ldb + nB + nseg), 0, 1);
      }
    }
    __syncthreads();

    // ---------------- fragments from LDS ----------------------------------
    // A 16x32 f16: lane half hh holds K = 16*hh..+15.
    const int kbA = hh * 16;
    v8h alo = *(const v8h*)&As[mwl + l16][kbA];
    v8h ahi = *(const v8h*)&As[mwl + l16][kbA + 8];
    v16h af;
#pragma unroll
    for (int j = 0; j < 8; ++j) { af[j] = alo[j]; af[8 + j] = ahi[j]; }

    // B 32x16 f16: lane half hh holds K = 16*hh..+15, column N = l16 (+0/+16).
    const int kbB = hh * 16;
    const _Float16* bp0 = &Bs[nwl + l16][kbB];
    const _Float16* bp1 = &Bs[nwl + 16 + l16][kbB];
    v8h b0a = *(const v8h*)bp0;
    v8h b0b = *(const v8h*)(bp0 + 8);
    v8h b1a = *(const v8h*)bp1;
    v8h b1b = *(const v8h*)(bp1 + 8);
    v16h bf0, bf1;
#pragma unroll
    for (int j = 0; j < 8; ++j) {
      bf0[j] = b0a[j]; bf0[8 + j] = b0b[j];
      bf1[j] = b1a[j]; bf1[8 + j] = b1b[j];
    }

    acc0 = __builtin_amdgcn_wmma_f32_16x16x32_f16(false, af, false, bf0,
                                                  (short)0, acc0, false, false);
    acc1 = __builtin_amdgcn_wmma_f32_16x16x32_f16(false, af, false, bf1,
                                                  (short)0, acc1, false, false);
    __syncthreads();
  }

  // ---- Store: C/D layout — VGPR r: M = r + 8*hh, N = l16.
  const int col0 = nB + nwl + l16;
  const int col1 = col0 + 16;
#pragma unroll
  for (int r = 0; r < 8; ++r) {
    const int row = mB + mwl + r + 8 * hh;
    float v0 = acc0[r] * alpha;
    float v1 = acc1[r] * alpha;
    if constexpr (EPI >= 1) {
      v0 += bias[col0];
      v1 += bias[col1];
    }
    if constexpr (EPI == 1) {
      v0 = fmaxf(v0, 0.0f);
      v1 = fmaxf(v1, 0.0f);
    }
    C[(long)row * ldc + col0] = v0;
    C[(long)row * ldc + col1] = v1;
  }
}

// ---------------------------------------------------------------------------
// x[b,s,d] = (emb[tok[b,s]][d] + pe(s,d)) * sqrt(D)
// pe(s,d) = sin/cos( s / 10000^(2d/D) ), sin for even d, cos for odd d.
// ---------------------------------------------------------------------------
__global__ __launch_bounds__(256) void embed_kernel(
    const int* __restrict__ tok, const float* __restrict__ emb,
    float* __restrict__ x)
{
  long i = (long)blockIdx.x * blockDim.x + threadIdx.x;
  const long tot = (long)B_ * S_ * D_;
  if (i >= tot) return;
  int d = (int)(i % D_);
  long bs = i / D_;
  int s = (int)(bs % S_);
  int t = tok[bs];
  float ang = (float)s * powf(10000.0f, -2.0f * (float)d / (float)D_);
  float pe = ((d & 1) == 0) ? sinf(ang) : cosf(ang);
  x[i] = (emb[(long)t * D_ + d] + pe) * 22.627416997969522f;  // sqrt(512)
}

// ---------------------------------------------------------------------------
// x = LN(x + h) with per-(s,d) gain/bias; std is biased (ddof=0):
//   out = g/(std+eps)*(x-m) + b.  One 256-thread block per row (D=512).
// ---------------------------------------------------------------------------
__global__ __launch_bounds__(256) void ln_residual_kernel(
    float* __restrict__ x, const float* __restrict__ h,
    const float* __restrict__ gain, const float* __restrict__ bias)
{
  const int T = 256;
  const long row = blockIdx.x;            // B*S rows
  const int s = (int)(row % S_);
  float* xr = x + row * D_;
  const float* hr = h + row * D_;
  const float* gr = gain + (long)s * D_;
  const float* br = bias + (long)s * D_;

  __shared__ float sm[T];
  const int tid = threadIdx.x;
  float v0 = xr[tid] + hr[tid];
  float v1 = xr[tid + T] + hr[tid + T];

  sm[tid] = v0 + v1;
  __syncthreads();
  for (int st = T / 2; st > 0; st >>= 1) {
    if (tid < st) sm[tid] += sm[tid + st];
    __syncthreads();
  }
  const float mean = sm[0] * (1.0f / (float)D_);
  __syncthreads();

  const float d0 = v0 - mean, d1 = v1 - mean;
  sm[tid] = d0 * d0 + d1 * d1;
  __syncthreads();
  for (int st = T / 2; st > 0; st >>= 1) {
    if (tid < st) sm[tid] += sm[tid + st];
    __syncthreads();
  }
  const float stdv = sqrtf(sm[0] * (1.0f / (float)D_));
  const float inv = 1.0f / (stdv + 1e-6f);

  xr[tid]     = gr[tid]     * inv * d0 + br[tid];
  xr[tid + T] = gr[tid + T] * inv * d1 + br[tid + T];
}

// ---------------------------------------------------------------------------
// In-place softmax over rows of length S (scores [B,H,S,S]); optional faithful
// "mask" adds 1e-15 to strictly-upper-triangle before softmax.
// ---------------------------------------------------------------------------
__global__ __launch_bounds__(256) void softmax_scores_kernel(
    float* __restrict__ sc, int masked)
{
  const int T = 256, PER = S_ / T;        // 4
  const long row = blockIdx.x;            // B*H*S rows
  const int s = (int)(row % S_);
  float* r = sc + row * (long)S_;
  __shared__ float sm[T];
  const int tid = threadIdx.x;

  float v[PER];
  float mx = -3.0e38f;
#pragma unroll
  for (int j = 0; j < PER; ++j) {
    const int t = tid + j * T;
    float val = r[t];
    if (masked && t > s) val += 1e-15f;
    v[j] = val;
    mx = fmaxf(mx, val);
  }
  sm[tid] = mx;
  __syncthreads();
  for (int st = T / 2; st > 0; st >>= 1) {
    if (tid < st) sm[tid] = fmaxf(sm[tid], sm[tid + st]);
    __syncthreads();
  }
  mx = sm[0];
  __syncthreads();

  float sum = 0.0f;
#pragma unroll
  for (int j = 0; j < PER; ++j) {
    v[j] = expf(v[j] - mx);
    sum += v[j];
  }
  sm[tid] = sum;
  __syncthreads();
  for (int st = T / 2; st > 0; st >>= 1) {
    if (tid < st) sm[tid] += sm[tid + st];
    __syncthreads();
  }
  const float inv = 1.0f / sm[0];
#pragma unroll
  for (int j = 0; j < PER; ++j) r[tid + j * T] = v[j] * inv;
}

// ---------------------------------------------------------------------------
// In-place softmax over rows of length V (output logits -> probs).
// ---------------------------------------------------------------------------
__global__ __launch_bounds__(256) void softmax_vocab_kernel(float* __restrict__ out)
{
  const int T = 256;
  const long row = blockIdx.x;            // B*S rows
  float* r = out + row * (long)V_;
  __shared__ float sm[T];
  const int tid = threadIdx.x;

  float mx = -3.0e38f;
  for (int t = tid; t < V_; t += T) mx = fmaxf(mx, r[t]);
  sm[tid] = mx;
  __syncthreads();
  for (int st = T / 2; st > 0; st >>= 1) {
    if (tid < st) sm[tid] = fmaxf(sm[tid], sm[tid + st]);
    __syncthreads();
  }
  mx = sm[0];
  __syncthreads();

  float sum = 0.0f;
  for (int t = tid; t < V_; t += T) {
    float e = expf(r[t] - mx);
    r[t] = e;
    sum += e;
  }
  sm[tid] = sum;
  __syncthreads();
  for (int st = T / 2; st > 0; st >>= 1) {
    if (tid < st) sm[tid] += sm[tid + st];
    __syncthreads();
  }
  const float inv = 1.0f / sm[0];
  for (int t = tid; t < V_; t += T) r[t] *= inv;
}

// ---------------------------------------------------------------------------
extern "C" void kernel_launch(void* const* d_in, const int* in_sizes, int n_in,
                              void* d_out, int out_size, void* d_ws, size_t ws_size,
                              hipStream_t stream)
{
  (void)in_sizes; (void)n_in; (void)out_size; (void)ws_size;
  const int*   tok = (const int*)  d_in[0];
  const float* emb = (const float*)d_in[1];
  const float* Wq  = (const float*)d_in[2];   // [L,2,H,D,DK]
  const float* Wk  = (const float*)d_in[3];
  const float* Wv  = (const float*)d_in[4];
  const float* Wo  = (const float*)d_in[5];   // [L,2,D,D]
  const float* lng = (const float*)d_in[6];   // [L,3,S,D]
  const float* lnb = (const float*)d_in[7];
  const float* fw1 = (const float*)d_in[8];   // [L,D,D]
  const float* fb1 = (const float*)d_in[9];   // [L,D]
  const float* fw2 = (const float*)d_in[10];
  const float* fb2 = (const float*)d_in[11];
  float* out = (float*)d_out;                 // [B,S,V] fp32 probs

  // Workspace carve (all fp32). Needs ~96 MB.
  char* p = (char*)d_ws;
  auto carve = [&](size_t nfloats) { float* q = (float*)p; p += nfloats * sizeof(float); return q; };
  const size_t NBSD = (size_t)B_ * S_ * D_;            // 1,048,576
  float* x  = carve(NBSD);
  float* qb = carve(NBSD);
  float* kb = carve(NBSD);
  float* vb = carve(NBSD);
  float* ob = carve(NBSD);
  float* hb = carve(NBSD);
  float* tb = carve(NBSD);
  float* sc = carve((size_t)B_ * H_ * S_ * S_);        // 16,777,216

  const dim3 blk(256);
  const int  BS = B_ * S_;                             // 2048 rows

  // 1) Embedding + positional encoding
  {
    long tot = (long)B_ * S_ * D_;
    embed_kernel<<<dim3((unsigned)((tot + 255) / 256)), blk, 0, stream>>>(tok, emb, x);
  }

  const long WQKV_sub = (long)H_ * D_ * DK_;           // per (l,i) slab
  const long WOO_sub  = (long)D_ * D_;

  for (int l = 0; l < L_; ++l) {
    for (int i = 0; i < 2; ++i) {
      const float* wq = Wq + ((long)l * 2 + i) * WQKV_sub;
      const float* wk = Wk + ((long)l * 2 + i) * WQKV_sub;
      const float* wv = Wv + ((long)l * 2 + i) * WQKV_sub;
      const float* wo = Wo + ((long)l * 2 + i) * WOO_sub;

      // --- Q/K/V projections: batched over heads. M=BS, N=DK, K=D. NN.
      // Output col = h*DK + k (concat-heads layout), ldc = D.
      dim3 gqkv(1, BS / 64, H_);
      gemm_wmma<false, 0><<<gqkv, blk, 0, stream>>>(
          x, wq, qb, nullptr, BS, DK_, D_, D_, DK_, D_,
          H_, 0, 0, 0, (long)D_ * DK_, 0, (long)DK_, 1.0f);
      gemm_wmma<false, 0><<<gqkv, blk, 0, stream>>>(
          x, wk, kb, nullptr, BS, DK_, D_, D_, DK_, D_,
          H_, 0, 0, 0, (long)D_ * DK_, 0, (long)DK_, 1.0f);
      gemm_wmma<false, 0><<<gqkv, blk, 0, stream>>>(
          x, wv, vb, nullptr, BS, DK_, D_, D_, DK_, D_,
          H_, 0, 0, 0, (long)D_ * DK_, 0, (long)DK_, 1.0f);

      // --- scores = (q @ k^T) / sqrt(DK): per (b,h). M=N=S, K=DK. NT.
      dim3 gsc(S_ / 64, S_ / 64, B_ * H_);
      gemm_wmma<true, 0><<<gsc, blk, 0, stream>>>(
          qb, kb, sc, nullptr, S_, S_, DK_, D_, D_, S_,
          H_, (long)S_ * D_, (long)DK_, (long)S_ * D_, (long)DK_,
          (long)H_ * S_ * S_, (long)S_ * S_, 0.125f /* 1/sqrt(64) */);

      // --- softmax rows (masked variant for first MHA of each layer)
      softmax_scores_kernel<<<dim3((unsigned)(B_ * H_ * S_)), blk, 0, stream>>>(
          sc, (i == 0) ? 1 : 0);

      // --- o = attn @ v: per (b,h). M=S, N=DK, K=S. NN.
      dim3 gav(1, S_ / 64, B_ * H_);
      gemm_wmma<false, 0><<<gav, blk, 0, stream>>>(
          sc, vb, ob, nullptr, S_, DK_, S_, S_, D_, D_,
          H_, (long)H_ * S_ * S_, (long)S_ * S_, (long)S_ * D_, (long)DK_,
          (long)S_ * D_, (long)DK_, 1.0f);

      // --- output projection: h = o @ Wo. M=BS, N=D, K=D. NN.
      dim3 gpo(D_ / 64, BS / 64, 1);
      gemm_wmma<false, 0><<<gpo, blk, 0, stream>>>(
          ob, wo, hb, nullptr, BS, D_, D_, D_, D_, D_,
          1, 0, 0, 0, 0, 0, 0, 1.0f);

      // --- x = LN(x + h), gain/bias slab j = i (0 or 1)
      const long lnoff = ((long)l * 3 + i) * S_ * D_;
      ln_residual_kernel<<<dim3((unsigned)BS), blk, 0, stream>>>(
          x, hb, lng + lnoff, lnb + lnoff);
    }

    // --- FFN: t = relu(x@w1 + b1); h = t@w2 + b2; x = LN(x + h)
    {
      dim3 gff(D_ / 64, BS / 64, 1);
      gemm_wmma<false, 1><<<gff, blk, 0, stream>>>(
          x, fw1 + (long)l * D_ * D_, tb, fb1 + (long)l * D_,
          BS, D_, D_, D_, D_, D_, 1, 0, 0, 0, 0, 0, 0, 1.0f);
      gemm_wmma<false, 2><<<gff, blk, 0, stream>>>(
          tb, fw2 + (long)l * D_ * D_, hb, fb2 + (long)l * D_,
          BS, D_, D_, D_, D_, D_, 1, 0, 0, 0, 0, 0, 0, 1.0f);
      const long lnoff = ((long)l * 3 + 2) * S_ * D_;
      ln_residual_kernel<<<dim3((unsigned)BS), blk, 0, stream>>>(
          x, hb, lng + lnoff, lnb + lnoff);
    }
  }

  // --- Tied output projection: logits = x @ emb^T (NT, N=V), then softmax.
  {
    dim3 gv(V_ / 64, BS / 64, 1);
    gemm_wmma<true, 0><<<gv, blk, 0, stream>>>(
        x, emb, out, nullptr, BS, V_, D_, D_, D_, V_,
        1, 0, 0, 0, 0, 0, 0, 1.0f);
    softmax_vocab_kernel<<<dim3((unsigned)BS), blk, 0, stream>>>(out);
  }
}
